// Fingerprint_viz_33182917329367
// MI455X (gfx1250) — matrix operand order repaired and checked by reading, not compiled, not run on hardware
//
#include <hip/hip_runtime.h>
#include <hip/hip_bf16.h>
#include <math.h>

typedef _Float16 f16;
typedef __attribute__((ext_vector_type(8)))  _Float16 v8h;
typedef __attribute__((ext_vector_type(16))) _Float16 v16h;
typedef __attribute__((ext_vector_type(8)))  float    v8f;

#define Bv 512
#define Lv 128
#define Kv 6
#define AFv 39
#define BFv 10
#define FPv 200
#define Rv 3
#define Tv 2
#define Mv (Bv*Lv)          // 65536
#define MNv (Bv*Lv*Kv)      // 393216
#define KP_S 64             // padded K for 39/49
#define KP_L 224            // padded K for 200
#define NP_FP 208           // padded N for 200
#define NP_3FP 608          // padded N for 600
#define NEGF (-9e8f)

__device__ __forceinline__ float leakyf(float x) { return x >= 0.f ? x : 0.01f * x; }
__device__ __forceinline__ float sigmf(float x)  { return 1.f / (1.f + __expf(-x)); }

__device__ __forceinline__ float wave_reduce_sum(float v) {
    #pragma unroll
    for (int m = 16; m >= 1; m >>= 1) v += __shfl_xor(v, m, 32);
    return v;
}

// ---------------------------------------------------------------------------
// WMMA GEMM on padded f16 staging: out = act( A[M,KP] @ W[Npad,KP]^T + rs*bias )
// Branch-free inner loop: 4x b128 loads + 1 wmma per 32-K step, fully unrolled.
// M must be a multiple of 16; W staging has >= ntilesN*16 zero-padded rows.
// ---------------------------------------------------------------------------
template<int KP>
__global__ void gemm_wmma(const f16* __restrict__ A, const f16* __restrict__ W,
                          const float* __restrict__ bias,
                          const float* __restrict__ rowscale,
                          float* __restrict__ out0,   // pre-act f32 (nullable)
                          float* __restrict__ out1,   // activated f32 (nullable)
                          f16*   __restrict__ outh,   // activated f16 (nullable)
                          int ldh,
                          int M, int N, int act, int ntilesN, int totalTiles)
{
    int wave = threadIdx.x >> 5;
    int tile = blockIdx.x * (blockDim.x >> 5) + wave;
    if (tile >= totalTiles) return;                   // wave-uniform
    int mt = tile / ntilesN, nt = tile % ntilesN;
    int lane = threadIdx.x & 31;
    int hf   = lane >> 4;
    int rc   = lane & 15;

    const f16* Ar = A + (size_t)(mt * 16 + rc) * KP;
    const f16* Wr = W + (size_t)(nt * 16 + rc) * KP;

    v8f c = {};
    #pragma unroll
    for (int k0 = 0; k0 < KP; k0 += 32) {
        v8h alo = *(const v8h*)(Ar + k0 + 8 * hf);
        v8h ahi = *(const v8h*)(Ar + k0 + 16 + 8 * hf);
        v8h blo = *(const v8h*)(Wr + k0 + 8 * hf);
        v8h bhi = *(const v8h*)(Wr + k0 + 16 + 8 * hf);
        v16h a = __builtin_shufflevector(alo, ahi, 0,1,2,3,4,5,6,7,8,9,10,11,12,13,14,15);
        v16h b = __builtin_shufflevector(blo, bhi, 0,1,2,3,4,5,6,7,8,9,10,11,12,13,14,15);
        c = __builtin_amdgcn_wmma_f32_16x16x32_f16(false, a, false, b,
                                                   (short)0, c, false, false);
    }

    int col = nt * 16 + rc;
    if (col < N) {
        float bv = bias ? bias[col] : 0.f;
        #pragma unroll
        for (int v = 0; v < 8; ++v) {
            int orow = mt * 16 + v + 8 * hf;
            float rs  = rowscale ? rowscale[orow] : 1.f;
            float val = c[v] + bv * rs;
            float av  = val;
            if (act == 1)      av = leakyf(val);
            else if (act == 2) av = val > 0.f ? val : expm1f(val);
            if (out0) out0[(size_t)orow * N + col] = val;
            if (out1) out1[(size_t)orow * N + col] = av;
            if (outh) outh[(size_t)orow * ldh + col] = (f16)av;
        }
    }
}

// f32 [N,K] -> zero-padded f16 [Npad,Kpad]
__global__ void cvt_w(const float* __restrict__ W, f16* __restrict__ dst,
                      int N, int K, int Npad, int Kpad)
{
    long long idx = (long long)blockIdx.x * blockDim.x + threadIdx.x;
    if (idx >= (long long)Npad * Kpad) return;
    int n = (int)(idx / Kpad), k = (int)(idx % Kpad);
    dst[idx] = (n < N && k < K) ? (f16)W[(size_t)n * K + k] : (f16)0.f;
}

// zero pad columns [startcol, ld) of an f16 [M, ld] staging buffer
__global__ void pad_zero16(f16* __restrict__ buf, int M, int ld, int startcol)
{
    int pw = ld - startcol;
    long long idx = (long long)blockIdx.x * blockDim.x + threadIdx.x;
    if (idx >= (long long)M * pw) return;
    int m = (int)(idx / pw), cc = startcol + (int)(idx % pw);
    buf[(size_t)m * ld + cc] = (f16)0.f;
}

// Gathered concat input, f16 padded: X[MN,64] = [atom_nb | bond_nb | 0...]
__global__ void build_neigh_input16(const float* __restrict__ atom_list,
                                    const float* __restrict__ bond_list,
                                    const int* __restrict__ adeg,
                                    const int* __restrict__ bdeg,
                                    f16* __restrict__ X)
{
    long long idx = (long long)blockIdx.x * blockDim.x + threadIdx.x;
    if (idx >= (long long)MNv * KP_S) return;
    int m = (int)(idx / KP_S), f = (int)(idx % KP_S);
    int b = m / (Lv * Kv);
    float v = 0.f;
    if (f < AFv) {
        int a = adeg[m];
        v = atom_list[((size_t)(b * Lv + a)) * AFv + f];
    } else if (f < AFv + BFv) {
        int bb = bdeg[m];
        v = bond_list[((size_t)(b * Lv + bb)) * BFv + (f - AFv)];
    }
    X[idx] = (f16)v;
}

// out[r] = dot(rows[r,:], vec) + bias; optional relu on input. One wave per row.
__global__ void rowdot(const float* __restrict__ rows, const float* __restrict__ vec,
                       float* __restrict__ out, const float* __restrict__ bias,
                       int M, int F, int relu_in)
{
    int g = (blockIdx.x * blockDim.x + threadIdx.x) >> 5;
    if (g >= M) return;
    int lane = threadIdx.x & 31;
    const float* rp = rows + (size_t)g * F;
    float s = 0.f;
    for (int f = lane; f < F; f += 32) {
        float x = rp[f];
        if (relu_in) x = fmaxf(x, 0.f);
        s += x * vec[f];
    }
    s = wave_reduce_sum(s);
    if (lane == 0) out[g] = s + (bias ? bias[0] : 0.f);
}

// Per (b,l): sn_k dots + leaky + pad mask + softmax over K=6. One wave per (b,l).
__global__ void atom_attn(const float* __restrict__ nsrc, int gmode,
                          const int* __restrict__ adeg,
                          const float* __restrict__ vec,
                          const float* __restrict__ alignb_p,
                          const float* __restrict__ sa,
                          float* __restrict__ w_out,
                          float* __restrict__ wsum_out)
{
    int bl = (blockIdx.x * blockDim.x + threadIdx.x) >> 5;
    if (bl >= Mv) return;
    int lane = threadIdx.x & 31;
    int b = bl >> 7;
    float sc[Kv]; bool pad[Kv];
    float sav = sa[bl], ab = alignb_p[0];
    #pragma unroll
    for (int k = 0; k < Kv; ++k) {
        int idx = adeg[(size_t)bl * Kv + k];
        pad[k] = (idx == Lv - 1);
        const float* rp = gmode ? (nsrc + ((size_t)(b * Lv + idx)) * FPv)
                                : (nsrc + ((size_t)bl * Kv + k) * FPv);
        float s = 0.f;
        for (int f = lane; f < FPv; f += 32) s += rp[f] * vec[f];
        s = wave_reduce_sum(s);
        sc[k] = leakyf(sav + s + ab) + (pad[k] ? NEGF : 0.f);
    }
    float mx = sc[0];
    #pragma unroll
    for (int k = 1; k < Kv; ++k) mx = fmaxf(mx, sc[k]);
    float e[Kv], esum = 0.f;
    #pragma unroll
    for (int k = 0; k < Kv; ++k) { e[k] = __expf(sc[k] - mx); esum += e[k]; }
    if (lane == 0) {
        float ws = 0.f;
        #pragma unroll
        for (int k = 0; k < Kv; ++k) {
            float w = (e[k] / esum) * (pad[k] ? 0.f : 1.f);
            w_out[(size_t)bl * Kv + k] = w;
            ws += w;
        }
        wsum_out[bl] = ws;
    }
}

// agg16[bl,f(<224)] = sum_k w[bl,k]*neigh_row_k[f] (f16, zero pad)
__global__ void weighted_agg16(const float* __restrict__ nsrc, int gmode,
                               const int* __restrict__ adeg,
                               const float* __restrict__ w,
                               f16* __restrict__ agg)
{
    long long idx = (long long)blockIdx.x * blockDim.x + threadIdx.x;
    if (idx >= (long long)Mv * KP_L) return;
    int bl = (int)(idx / KP_L), f = (int)(idx % KP_L);
    float s = 0.f;
    if (f < FPv) {
        int b = bl >> 7;
        #pragma unroll
        for (int k = 0; k < Kv; ++k) {
            float wk = w[(size_t)bl * Kv + k];
            const float* rp;
            if (gmode) {
                int a = adeg[(size_t)bl * Kv + k];
                rp = nsrc + ((size_t)(b * Lv + a)) * FPv;
            } else {
                rp = nsrc + ((size_t)bl * Kv + k) * FPv;
            }
            s += wk * rp[f];
        }
    }
    agg[idx] = (f16)s;
}

// GRU combine: h (f32) in place + f16 staging; optional outputs
__global__ void gru_combine(const float* __restrict__ gi, const float* __restrict__ gh,
                            float* __restrict__ h,
                            float* __restrict__ out_pre, float* __restrict__ out_act,
                            f16* __restrict__ h16, f16* __restrict__ act16,
                            int M)
{
    long long idx = (long long)blockIdx.x * blockDim.x + threadIdx.x;
    if (idx >= (long long)M * FPv) return;
    int m = (int)(idx / FPv), f = (int)(idx % FPv);
    const float* gim = gi + (size_t)m * 3 * FPv;
    const float* ghm = gh + (size_t)m * 3 * FPv;
    float r = sigmf(gim[f] + ghm[f]);
    float z = sigmf(gim[FPv + f] + ghm[FPv + f]);
    float n = tanhf(gim[2 * FPv + f] + r * ghm[2 * FPv + f]);
    float hv = h[idx];
    float hn = (1.f - z) * n + z * hv;
    h[idx] = hn;
    if (out_pre) out_pre[idx] = hn;
    if (out_act) out_act[idx] = fmaxf(hn, 0.f);
    if (h16)   h16[(size_t)m * KP_L + f] = (f16)hn;
    if (act16) act16[(size_t)m * KP_L + f] = (f16)fmaxf(hn, 0.f);
}

// dst[b,f] = sum_l src[b,l,f]*mask[b,l]; optional second f32 and padded f16 copy
__global__ void masked_sum(const float* __restrict__ src, const float* __restrict__ mask,
                           float* __restrict__ dst, float* __restrict__ dst2,
                           f16* __restrict__ dst16)
{
    int idx = blockIdx.x * blockDim.x + threadIdx.x;
    if (idx >= Bv * KP_L) return;
    int b = idx / KP_L, f = idx % KP_L;
    if (f < FPv) {
        float s = 0.f;
        for (int l = 0; l < Lv; ++l)
            s += src[((size_t)(b * Lv + l)) * FPv + f] * mask[b * Lv + l];
        dst[b * FPv + f] = s;
        if (dst2) dst2[b * FPv + f] = s;
        if (dst16) dst16[(size_t)b * KP_L + f] = (f16)s;
    } else if (dst16) {
        dst16[(size_t)b * KP_L + f] = (f16)0.f;
    }
}

// Per-molecule softmax over L=128 atoms
__global__ void mol_softmax(const float* __restrict__ sa, const float* __restrict__ sn,
                            const float* __restrict__ alignb_p,
                            const float* __restrict__ mask,
                            float* __restrict__ w_out)
{
    __shared__ float sh[Lv];
    int b = blockIdx.x, l = threadIdx.x;
    float mk = mask[b * Lv + l];
    float sc = leakyf(sa[b] + sn[b * Lv + l] + alignb_p[0]) + (mk == 0.f ? NEGF : 0.f);
    sh[l] = sc; __syncthreads();
    for (int s = Lv / 2; s > 0; s >>= 1) {
        if (l < s) sh[l] = fmaxf(sh[l], sh[l + s]);
        __syncthreads();
    }
    float mx = sh[0]; __syncthreads();
    float e = __expf(sc - mx);
    sh[l] = e; __syncthreads();
    for (int s = Lv / 2; s > 0; s >>= 1) {
        if (l < s) sh[l] += sh[l + s];
        __syncthreads();
    }
    w_out[b * Lv + l] = (e / sh[0]) * mk;
}

// mctx16[b,f] = elu( sum_l w[b,l]*at[b,l,f] ), f16 padded
__global__ void mol_ctx16(const float* __restrict__ w, const float* __restrict__ at,
                          f16* __restrict__ ctx)
{
    int idx = blockIdx.x * blockDim.x + threadIdx.x;
    if (idx >= Bv * KP_L) return;
    int b = idx / KP_L, f = idx % KP_L;
    float s = 0.f;
    if (f < FPv) {
        for (int l = 0; l < Lv; ++l)
            s += w[b * Lv + l] * at[((size_t)(b * Lv + l)) * FPv + f];
        s = s > 0.f ? s : expm1f(s);
    }
    ctx[idx] = (f16)s;
}

// ---------------------------------------------------------------------------
static inline int cdiv(long long a, long long b) { return (int)((a + b - 1) / b); }

template<int KP>
static void launch_gemm(const f16* A, const f16* W, const float* bias, const float* rs,
                        float* out0, float* out1, f16* outh, int ldh,
                        int M, int N, int act, hipStream_t s)
{
    int ntN = (N + 15) / 16;
    int tiles = (M / 16) * ntN;
    gemm_wmma<KP><<<cdiv(tiles, 8), 256, 0, s>>>(A, W, bias, rs, out0, out1, outh, ldh,
                                                 M, N, act, ntN, tiles);
}

extern "C" void kernel_launch(void* const* d_in, const int* in_sizes, int n_in,
                              void* d_out, int out_size, void* d_ws, size_t ws_size,
                              hipStream_t stream)
{
    (void)in_sizes; (void)n_in; (void)out_size; (void)ws_size;
    const float* atom_list = (const float*)d_in[0];
    const float* bond_list = (const float*)d_in[1];
    const int*   adeg      = (const int*)d_in[2];
    const int*   bdeg      = (const int*)d_in[3];
    const float* amask     = (const float*)d_in[4];
    const float* atom_fc_W = (const float*)d_in[5];
    const float* atom_fc_b = (const float*)d_in[6];
    const float* nb_fc_W   = (const float*)d_in[7];
    const float* nb_fc_b   = (const float*)d_in[8];
    const float* align_W   = (const float*)d_in[9];
    const float* align_b   = (const float*)d_in[10];
    const float* attend_W  = (const float*)d_in[11];
    const float* attend_b  = (const float*)d_in[12];
    const float* gru_Wih   = (const float*)d_in[13];
    const float* gru_Whh   = (const float*)d_in[14];
    const float* gru_bih   = (const float*)d_in[15];
    const float* gru_bhh   = (const float*)d_in[16];
    const float* mal_W     = (const float*)d_in[17];
    const float* mal_b     = (const float*)d_in[18];
    const float* mat_W     = (const float*)d_in[19];
    const float* mat_b     = (const float*)d_in[20];
    const float* mg_Wih    = (const float*)d_in[21];
    const float* mg_Whh    = (const float*)d_in[22];
    const float* mg_bih    = (const float*)d_in[23];
    const float* mg_bhh    = (const float*)d_in[24];
    const float* out_W     = (const float*)d_in[25];
    const float* out_b     = (const float*)d_in[26];

    float* out = (float*)d_out;

    // d_out layout
    float* o_atomviz = out;                                   // [4,M,FP]
    float* o_attn    = o_atomviz + (size_t)4 * Mv * FPv;      // [3,MN]
    float* o_molviz  = o_attn + (size_t)3 * MNv;              // [3,B,FP]
    float* o_molunb  = o_molviz + (size_t)3 * Bv * FPv;       // [3,B,FP]
    float* o_molattn = o_molunb + (size_t)3 * Bv * FPv;       // [2,B,L]
    float* o_pred    = o_molattn + (size_t)2 * Bv * Lv;       // [B]

    // ---- f32 workspace ----
    float* wsf   = (float*)d_ws;
    float* w_neigh = wsf;                                     // [MN,FP]
    float* w_gi    = w_neigh;                                 // alias (neigh dead first)
    float* w_gh    = w_neigh + (size_t)Mv * 3 * FPv;          // alias
    float* w_h     = w_neigh + (size_t)MNv * FPv;             // [M,FP]
    float* w_at    = w_h + (size_t)Mv * FPv;                  // [M,FP]
    float* w_sa    = w_at + (size_t)Mv * FPv;                 // [M]
    float* w_wsum  = w_sa + Mv;                               // [M]
    float* w_snmol = w_wsum + Mv;                             // [B*L]
    float* w_samol = w_snmol + (size_t)Bv * Lv;               // [B]
    float* w_mfeat = w_samol + Bv;                            // [B,FP]
    float* w_mgi   = w_mfeat + (size_t)Bv * FPv;              // [B,600]
    float* w_mgh   = w_mgi + (size_t)Bv * 3 * FPv;            // [B,600]
    float* w_f32end = w_mgh + (size_t)Bv * 3 * FPv;

    // ---- f16 workspace (16B aligned: all sizes multiples of 16) ----
    f16* h16     = (f16*)w_f32end;
    f16* X16     = h16 + (size_t)Mv * KP_L;                   // [MN,64]
    f16* atomA16 = X16 + (size_t)MNv * KP_S;                  // [M,64]
    f16* agg16   = atomA16 + (size_t)Mv * KP_S;               // [M,224]
    f16* ctx16   = agg16 + (size_t)Mv * KP_L;                 // [M,224]
    f16* act16   = ctx16 + (size_t)Mv * KP_L;                 // [M,224]
    f16* mfeat16 = act16 + (size_t)Mv * KP_L;                 // [B,224]
    f16* mctx16  = mfeat16 + (size_t)Bv * KP_L;               // [B,224]
    f16* wa_atom = mctx16 + (size_t)Bv * KP_L;                // [208,64]
    f16* wa_nb   = wa_atom + (size_t)NP_FP * KP_S;
    f16* wa_att  = wa_nb + (size_t)NP_FP * KP_S;              // [3][208,224]
    f16* wa_gih  = wa_att + (size_t)3 * NP_FP * KP_L;         // [3][608,224]
    f16* wa_ghh  = wa_gih + (size_t)3 * NP_3FP * KP_L;
    f16* wa_mat  = wa_ghh + (size_t)3 * NP_3FP * KP_L;        // [208,224]
    f16* wa_mgih = wa_mat + (size_t)NP_FP * KP_L;             // [608,224]
    f16* wa_mghh = wa_mgih + (size_t)NP_3FP * KP_L;

    #define CVT(src, dst, N_, K_, NP_, KP_) \
        cvt_w<<<cdiv((long long)(NP_) * (KP_), 256), 256, 0, stream>>>(src, dst, N_, K_, NP_, KP_)

    // weight + input staging
    CVT(atom_fc_W, wa_atom, FPv, AFv, NP_FP, KP_S);
    CVT(nb_fc_W,   wa_nb,   FPv, AFv + BFv, NP_FP, KP_S);
    for (int r = 0; r < Rv; ++r) {
        CVT(attend_W + (size_t)r * FPv * FPv, wa_att + (size_t)r * NP_FP * KP_L,
            FPv, FPv, NP_FP, KP_L);
        CVT(gru_Wih + (size_t)r * 3 * FPv * FPv, wa_gih + (size_t)r * NP_3FP * KP_L,
            3 * FPv, FPv, NP_3FP, KP_L);
        CVT(gru_Whh + (size_t)r * 3 * FPv * FPv, wa_ghh + (size_t)r * NP_3FP * KP_L,
            3 * FPv, FPv, NP_3FP, KP_L);
    }
    CVT(mat_W,  wa_mat,  FPv, FPv, NP_FP, KP_L);
    CVT(mg_Wih, wa_mgih, 3 * FPv, FPv, NP_3FP, KP_L);
    CVT(mg_Whh, wa_mghh, 3 * FPv, FPv, NP_3FP, KP_L);
    CVT(atom_list, atomA16, Mv, AFv, Mv, KP_S);

    // zero pad columns of persistent stagings
    pad_zero16<<<cdiv((long long)Mv * (KP_L - FPv), 256), 256, 0, stream>>>(h16,   Mv, KP_L, FPv);
    pad_zero16<<<cdiv((long long)Mv * (KP_L - FPv), 256), 256, 0, stream>>>(ctx16, Mv, KP_L, FPv);
    pad_zero16<<<cdiv((long long)Mv * (KP_L - FPv), 256), 256, 0, stream>>>(act16, Mv, KP_L, FPv);

    // atom_fc: pre -> atom_viz[0]; leaky -> h (f32) and h16
    launch_gemm<KP_S>(atomA16, wa_atom, atom_fc_b, nullptr,
                      o_atomviz, w_h, h16, KP_L, Mv, FPv, 1, stream);

    // neighbor_fc
    build_neigh_input16<<<cdiv((long long)MNv * KP_S, 256), 256, 0, stream>>>(
        atom_list, bond_list, adeg, bdeg, X16);
    launch_gemm<KP_S>(X16, wa_nb, nb_fc_b, nullptr,
                      nullptr, w_neigh, nullptr, 0, MNv, FPv, 1, stream);

    for (int r = 0; r < Rv; ++r) {
        const float* cur  = (r == 0) ? w_h : (o_atomviz + (size_t)r * Mv * FPv);
        const float* nsrc = (r == 0) ? w_neigh : (o_atomviz + (size_t)r * Mv * FPv);
        int gmode = (r == 0) ? 0 : 1;
        float* w_r = o_attn + (size_t)r * MNv;

        rowdot<<<cdiv((long long)Mv * 32, 256), 256, 0, stream>>>(
            cur, align_W + (size_t)r * 2 * FPv, w_sa, nullptr, Mv, FPv, 0);
        atom_attn<<<cdiv((long long)Mv * 32, 256), 256, 0, stream>>>(
            nsrc, gmode, adeg, align_W + (size_t)r * 2 * FPv + FPv,
            align_b + r, w_sa, w_r, w_wsum);
        weighted_agg16<<<cdiv((long long)Mv * KP_L, 256), 256, 0, stream>>>(
            nsrc, gmode, adeg, w_r, agg16);
        // ctx = elu(agg @ attend_W^T + wsum*attend_b) -> ctx16 only
        launch_gemm<KP_L>(agg16, wa_att + (size_t)r * NP_FP * KP_L,
                          attend_b + (size_t)r * FPv, w_wsum,
                          nullptr, nullptr, ctx16, KP_L, Mv, FPv, 2, stream);
        // GRU gates (f32 gi/gh alias dead neigh buffer)
        launch_gemm<KP_L>(ctx16, wa_gih + (size_t)r * NP_3FP * KP_L,
                          gru_bih + (size_t)r * 3 * FPv, nullptr,
                          w_gi, nullptr, nullptr, 0, Mv, 3 * FPv, 0, stream);
        launch_gemm<KP_L>(h16, wa_ghh + (size_t)r * NP_3FP * KP_L,
                          gru_bhh + (size_t)r * 3 * FPv, nullptr,
                          w_gh, nullptr, nullptr, 0, Mv, 3 * FPv, 0, stream);
        gru_combine<<<cdiv((long long)Mv * FPv, 256), 256, 0, stream>>>(
            w_gi, w_gh, w_h, nullptr, o_atomviz + (size_t)(r + 1) * Mv * FPv,
            h16, (r == Rv - 1) ? act16 : nullptr, Mv);
    }

    const float* act_last = o_atomviz + (size_t)3 * Mv * FPv;

    masked_sum<<<cdiv(Bv * KP_L, 256), 256, 0, stream>>>(w_h, amask, o_molunb, nullptr, nullptr);
    masked_sum<<<cdiv(Bv * KP_L, 256), 256, 0, stream>>>(act_last, amask, o_molviz, w_mfeat, mfeat16);

    rowdot<<<cdiv((long long)Mv * 32, 256), 256, 0, stream>>>(
        act_last, mal_W + FPv, w_snmol, nullptr, Mv, FPv, 0);
    launch_gemm<KP_L>(act16, wa_mat, mat_b, nullptr,
                      w_at, nullptr, nullptr, 0, Mv, FPv, 0, stream);

    for (int t = 0; t < Tv; ++t) {
        rowdot<<<cdiv(Bv * 32, 256), 256, 0, stream>>>(
            w_mfeat, mal_W, w_samol, nullptr, Bv, FPv, 1);
        float* wt = o_molattn + (size_t)t * Bv * Lv;
        mol_softmax<<<Bv, Lv, 0, stream>>>(w_samol, w_snmol, mal_b, amask, wt);
        mol_ctx16<<<cdiv(Bv * KP_L, 256), 256, 0, stream>>>(wt, w_at, mctx16);
        launch_gemm<KP_L>(mctx16, wa_mgih, mg_bih, nullptr,
                          w_mgi, nullptr, nullptr, 0, Bv, 3 * FPv, 0, stream);
        launch_gemm<KP_L>(mfeat16, wa_mghh, mg_bhh, nullptr,
                          w_mgh, nullptr, nullptr, 0, Bv, 3 * FPv, 0, stream);
        gru_combine<<<cdiv((long long)Bv * FPv, 256), 256, 0, stream>>>(
            w_mgi, w_mgh, w_mfeat,
            o_molunb + (size_t)(t + 1) * Bv * FPv,
            o_molviz + (size_t)(t + 1) * Bv * FPv,
            mfeat16, nullptr, Bv);
    }

    rowdot<<<cdiv(Bv * 32, 256), 256, 0, stream>>>(
        w_mfeat, out_W, o_pred, out_b, Bv, FPv, 0);
}
